// HybridDecoder_80272938762897
// MI455X (gfx1250) — compile-verified, matrix operand order checked by reading, not run on hardware
//
#include <hip/hip_runtime.h>
#include <math.h>

#define B_   32
#define T_   512
#define L_   128
#define D_   512
#define H_   8
#define HD_  64
#define NL_  6
#define DFF_ 1024
#define V_   8000

typedef __attribute__((ext_vector_type(16))) _Float16 v16h;
typedef __attribute__((ext_vector_type(8)))  _Float16 v8h;
typedef __attribute__((ext_vector_type(4)))  _Float16 v4h;
typedef __attribute__((ext_vector_type(8)))  float    v8f;

// ---------------------------------------------------------------------------
// WMMA fragment load: operand stored row-major [16 rows][K contiguous], ld in
// halves. Per ISA 7.12.2 (16-bit A 16x32): lane l (r=l&15, hi=l>>4) holds
// halves K = hi*8 + [0,8) and 16 + hi*8 + [0,8).  B ([N,K] row-major) is the
// mirror layout with n across lanes, so the same loader serves both.
// ---------------------------------------------------------------------------
__device__ __forceinline__ v16h load_frag(const _Float16* base, int ld) {
  const int lane = threadIdx.x & 31;
  const int r  = lane & 15;
  const int hi = lane >> 4;
  const _Float16* p = base + r * ld + hi * 8;
  v8h lo = *(const v8h*)(p);
  v8h hh = *(const v8h*)(p + 16);
  v16h out;
#pragma unroll
  for (int i = 0; i < 8; ++i) { out[i] = lo[i]; out[i + 8] = hh[i]; }
  return out;
}

// ---------------------------------------------------------------------------
// Generic batched GEMM:  C[bz] = A[bz] @ Bw[bz]^T  (+bias) (+exact GELU)
//   A : [M,K] row-major (lda), Bw : [N,K] row-major (ldb), C : [M,N] (ldc)
// Block tile 128x64, 8 waves, each wave 32x32 via 2x2 WMMA f32_16x16x32_f16.
// f32 inputs are converted to f16 while staging into LDS.
// Requires: M%128==0, N%64==0, K%32==0 (true for every GEMM in this model).
// ---------------------------------------------------------------------------
#define BM 128
#define BN 64
#define BK 32
#define LDSP 40   // padded LDS row length in halves (80B rows, 16B aligned)

__global__ __launch_bounds__(256) void gemm_xwt(
    const float* __restrict__ A, const float* __restrict__ Bw,
    const float* __restrict__ bias, float* __restrict__ C,
    int K, int lda, int ldb, int ldc,
    long long sA, long long sB, long long sC, int act)
{
  __shared__ _Float16 As[BM][LDSP];
  __shared__ _Float16 Bs[BN][LDSP];

  A  += (long long)blockIdx.z * sA;
  Bw += (long long)blockIdx.z * sB;
  C  += (long long)blockIdx.z * sC;

  const int tid  = threadIdx.x;
  const int wave = tid >> 5;
  const int wm   = wave >> 1;      // 0..3
  const int wn   = wave & 1;       // 0..1
  const int mbase = blockIdx.y * BM;
  const int nbase = blockIdx.x * BN;

  v8f acc00 = {}, acc01 = {}, acc10 = {}, acc11 = {};

  const int c4 = (tid & 7) * 4;    // float column within BK
  const int r0 = tid >> 3;         // 0..31

  for (int k0 = 0; k0 < K; k0 += BK) {
#pragma unroll
    for (int it = 0; it < 4; ++it) {            // A tile: 128 x 32
      int r = r0 + it * 32;
      const float4 f = *(const float4*)(A + (long long)(mbase + r) * lda + k0 + c4);
      v4h h; h[0] = (_Float16)f.x; h[1] = (_Float16)f.y;
             h[2] = (_Float16)f.z; h[3] = (_Float16)f.w;
      *(v4h*)&As[r][c4] = h;
    }
#pragma unroll
    for (int it = 0; it < 2; ++it) {            // B tile: 64 x 32
      int r = r0 + it * 32;
      const float4 f = *(const float4*)(Bw + (long long)(nbase + r) * ldb + k0 + c4);
      v4h h; h[0] = (_Float16)f.x; h[1] = (_Float16)f.y;
             h[2] = (_Float16)f.z; h[3] = (_Float16)f.w;
      *(v4h*)&Bs[r][c4] = h;
    }
    __syncthreads();

    v16h a0 = load_frag(&As[wm * 32 +  0][0], LDSP);
    v16h a1 = load_frag(&As[wm * 32 + 16][0], LDSP);
    v16h b0 = load_frag(&Bs[wn * 32 +  0][0], LDSP);
    v16h b1 = load_frag(&Bs[wn * 32 + 16][0], LDSP);

    acc00 = __builtin_amdgcn_wmma_f32_16x16x32_f16(false, a0, false, b0, (short)0, acc00, false, false);
    acc01 = __builtin_amdgcn_wmma_f32_16x16x32_f16(false, a0, false, b1, (short)0, acc01, false, false);
    acc10 = __builtin_amdgcn_wmma_f32_16x16x32_f16(false, a1, false, b0, (short)0, acc10, false, false);
    acc11 = __builtin_amdgcn_wmma_f32_16x16x32_f16(false, a1, false, b1, (short)0, acc11, false, false);
    __syncthreads();
  }

  const int lane = tid & 31;
  const int ln = lane & 15;
  const int hi = lane >> 4;
#pragma unroll
  for (int i = 0; i < 2; ++i) {
#pragma unroll
    for (int j = 0; j < 2; ++j) {
      const v8f acc = (i == 0) ? (j == 0 ? acc00 : acc01)
                               : (j == 0 ? acc10 : acc11);
      const int col = nbase + wn * 32 + j * 16 + ln;
      const float bv = bias ? bias[col] : 0.0f;
#pragma unroll
      for (int r = 0; r < 8; ++r) {
        const int row = mbase + wm * 32 + i * 16 + hi * 8 + r;
        float v = acc[r] + bv;
        if (act == 1) v = 0.5f * v * (1.0f + erff(v * 0.70710678118654752f));
        C[(long long)row * ldc + col] = v;
      }
    }
  }
}

// ---------------------------------------------------------------------------
// Elementwise / reduction kernels
// ---------------------------------------------------------------------------
__global__ void embed_k(const int* __restrict__ ids, const float* __restrict__ emb,
                        float* __restrict__ tgt) {
  long long idx = (long long)blockIdx.x * blockDim.x + threadIdx.x;   // B*L*D
  int d = idx & (D_ - 1);
  long long bl = idx >> 9;
  tgt[idx] = emb[(long long)ids[bl] * D_ + d] * 22.62741699796952f;   // sqrt(512)
}

__global__ void pad_mask_k(const float* __restrict__ mem, float* __restrict__ pad) {
  __shared__ float red[256];
  const int row = blockIdx.x;                    // B*T rows of D
  const int t = threadIdx.x;
  const float* m = mem + (long long)row * D_;
  red[t] = fabsf(m[t]) + fabsf(m[t + 256]);
  __syncthreads();
  for (int off = 128; off > 0; off >>= 1) {
    if (t < off) red[t] += red[t + off];
    __syncthreads();
  }
  if (t == 0) pad[row] = (red[0] == 0.0f) ? 1.0f : 0.0f;
}

// RoPE + head-split for self-attention:  Qh,Kh: [B,H,L,HD]; Vt: [B,H,HD,L]
__global__ void rope_split_k(const float* __restrict__ q, const float* __restrict__ k,
                             const float* __restrict__ v, float* __restrict__ Qh,
                             float* __restrict__ Kh, float* __restrict__ Vt) {
  long long idx = (long long)blockIdx.x * blockDim.x + threadIdx.x;   // B*L*D
  int d = idx & (D_ - 1);
  long long bl = idx >> 9;
  int l = (int)(bl & (L_ - 1));
  int b = (int)(bl >> 7);
  int h = d >> 6, hd = d & 63;
  int i = hd & 31;
  float inv = __powf(10000.0f, -(float)(2 * i) / 64.0f);
  float s, c;
  __sincosf((float)l * inv, &s, &c);
  float qv = q[idx], kv = k[idx];
  float qo = (hd < 32) ? -q[idx + 32] : q[idx - 32];
  float ko = (hd < 32) ? -k[idx + 32] : k[idx - 32];
  long long o = (((long long)(b * H_ + h) * L_ + l) * HD_) + hd;
  Qh[o] = qv * c + qo * s;
  Kh[o] = kv * c + ko * s;
  Vt[(((long long)(b * H_ + h) * HD_ + hd) * L_) + l] = v[idx];
}

__global__ void split_q_k(const float* __restrict__ q, float* __restrict__ Qh) {
  long long idx = (long long)blockIdx.x * blockDim.x + threadIdx.x;   // B*L*D
  int d = idx & (D_ - 1);
  long long bl = idx >> 9;
  int l = (int)(bl & (L_ - 1));
  int b = (int)(bl >> 7);
  int h = d >> 6, hd = d & 63;
  Qh[(((long long)(b * H_ + h) * L_ + l) * HD_) + hd] = q[idx];
}

// memory K/V head-split: K2h [B,H,T,HD]; V2t [B,H,HD,T]
__global__ void split_kv_mem_k(const float* __restrict__ kc, const float* __restrict__ vc,
                               float* __restrict__ K2h, float* __restrict__ V2t) {
  long long idx = (long long)blockIdx.x * blockDim.x + threadIdx.x;   // B*T*D
  int d = idx & (D_ - 1);
  long long bt = idx >> 9;
  int t = (int)(bt & (T_ - 1));
  int b = (int)(bt >> 9);
  int h = d >> 6, hd = d & 63;
  K2h[(((long long)(b * H_ + h) * T_ + t) * HD_) + hd] = kc[idx];
  V2t[(((long long)(b * H_ + h) * HD_ + hd) * T_) + t] = vc[idx];
}

__global__ void merge_heads_k(const float* __restrict__ Oh, float* __restrict__ y) {
  long long idx = (long long)blockIdx.x * blockDim.x + threadIdx.x;   // B*L*D
  int d = idx & (D_ - 1);
  long long bl = idx >> 9;
  int l = (int)(bl & (L_ - 1));
  int b = (int)(bl >> 7);
  int h = d >> 6, hd = d & 63;
  y[idx] = Oh[(((long long)(b * H_ + h) * L_ + l) * HD_) + hd];
}

// causal softmax, in place on S [B*H*L rows, L cols], scale 1/sqrt(HD)
__global__ void softmax_self_k(float* __restrict__ S) {
  __shared__ float red[128];
  const int row = blockIdx.x;
  const int l = row & (L_ - 1);
  float* s = S + (long long)row * L_;
  const int t = threadIdx.x;
  float v = s[t] * 0.125f;
  const bool ok = (t <= l);
  red[t] = ok ? v : -3.4e38f;
  __syncthreads();
  for (int off = 64; off > 0; off >>= 1) {
    if (t < off) red[t] = fmaxf(red[t], red[t + off]);
    __syncthreads();
  }
  const float m = red[0];
  __syncthreads();
  const float e = ok ? __expf(v - m) : 0.0f;
  red[t] = e;
  __syncthreads();
  for (int off = 64; off > 0; off >>= 1) {
    if (t < off) red[t] += red[t + off];
    __syncthreads();
  }
  s[t] = e / red[0];
}

// key-padding softmax, in place on S [B*H*L rows, T cols]
__global__ void softmax_cross_k(float* __restrict__ S, const float* __restrict__ pad) {
  __shared__ float red[256];
  const int row = blockIdx.x;
  const int b = row / (H_ * L_);
  float* s = S + (long long)row * T_;
  const int t = threadIdx.x;
  float v0 = s[t] * 0.125f, v1 = s[t + 256] * 0.125f;
  const bool m0 = (pad[b * T_ + t] != 0.0f);
  const bool m1 = (pad[b * T_ + t + 256] != 0.0f);
  red[t] = fmaxf(m0 ? -3.4e38f : v0, m1 ? -3.4e38f : v1);
  __syncthreads();
  for (int off = 128; off > 0; off >>= 1) {
    if (t < off) red[t] = fmaxf(red[t], red[t + off]);
    __syncthreads();
  }
  const float mx = red[0];
  __syncthreads();
  const float e0 = m0 ? 0.0f : __expf(v0 - mx);
  const float e1 = m1 ? 0.0f : __expf(v1 - mx);
  red[t] = e0 + e1;
  __syncthreads();
  for (int off = 128; off > 0; off >>= 1) {
    if (t < off) red[t] += red[t + off];
    __syncthreads();
  }
  const float inv = (red[0] > 0.0f) ? 1.0f / red[0] : 0.0f;
  s[t] = e0 * inv;
  s[t + 256] = e1 * inv;
}

// tgt = LayerNorm(tgt + y) * g + b, in place on tgt. Row length D=512.
__global__ void add_ln_k(float* __restrict__ tgt, const float* __restrict__ y,
                         const float* __restrict__ g, const float* __restrict__ bb) {
  __shared__ float red[256];
  const int row = blockIdx.x;
  const int t = threadIdx.x;
  float* xr = tgt + (long long)row * D_;
  const float* yr = y + (long long)row * D_;
  const float x0 = xr[t] + yr[t];
  const float x1 = xr[t + 256] + yr[t + 256];
  red[t] = x0 + x1;
  __syncthreads();
  for (int off = 128; off > 0; off >>= 1) {
    if (t < off) red[t] += red[t + off];
    __syncthreads();
  }
  const float mean = red[0] * (1.0f / D_);
  __syncthreads();
  const float d0 = x0 - mean, d1 = x1 - mean;
  red[t] = d0 * d0 + d1 * d1;
  __syncthreads();
  for (int off = 128; off > 0; off >>= 1) {
    if (t < off) red[t] += red[t + off];
    __syncthreads();
  }
  const float inv = rsqrtf(red[0] * (1.0f / D_) + 1e-5f);
  xr[t]       = d0 * inv * g[t]       + bb[t];
  xr[t + 256] = d1 * inv * g[t + 256] + bb[t + 256];
}

// ---------------------------------------------------------------------------
static inline void gemm(hipStream_t st, const float* A, const float* Bw,
                        const float* bias, float* C, int M, int N, int K,
                        int lda, int ldb, int ldc,
                        long long sA, long long sB, long long sC,
                        int batch, int act) {
  dim3 grid(N / BN, M / BM, batch);
  gemm_xwt<<<grid, dim3(256), 0, st>>>(A, Bw, bias, C, K, lda, ldb, ldc, sA, sB, sC, act);
}

extern "C" void kernel_launch(void* const* d_in, const int* in_sizes, int n_in,
                              void* d_out, int out_size, void* d_ws, size_t ws_size,
                              hipStream_t stream) {
  (void)in_sizes; (void)n_in; (void)out_size; (void)ws_size;
  const float* enc    = (const float*)d_in[0];
  const int*   tgids  = (const int*)  d_in[1];
  const float* emb    = (const float*)d_in[2];
  const float* inp_w  = (const float*)d_in[3];
  const float* inp_b  = (const float*)d_in[4];
  const float* out_w  = (const float*)d_in[5];
  const float* out_b  = (const float*)d_in[6];
  const float* qw     = (const float*)d_in[7];
  const float* qb     = (const float*)d_in[8];
  const float* kw     = (const float*)d_in[9];
  const float* kb     = (const float*)d_in[10];
  const float* vw     = (const float*)d_in[11];
  const float* vb     = (const float*)d_in[12];
  const float* ow     = (const float*)d_in[13];
  const float* ob     = (const float*)d_in[14];
  const float* mha_w  = (const float*)d_in[15];
  const float* mha_b  = (const float*)d_in[16];
  const float* mha_ow = (const float*)d_in[17];
  const float* mha_ob = (const float*)d_in[18];
  const float* w1     = (const float*)d_in[19];
  const float* b1     = (const float*)d_in[20];
  const float* w2     = (const float*)d_in[21];
  const float* b2     = (const float*)d_in[22];
  const float* ln1_g  = (const float*)d_in[23];
  const float* ln1_b  = (const float*)d_in[24];
  const float* ln2_g  = (const float*)d_in[25];
  const float* ln2_b  = (const float*)d_in[26];
  const float* ln3_g  = (const float*)d_in[27];
  const float* ln3_b  = (const float*)d_in[28];

  const long long BT = (long long)B_ * T_;      // 16384
  const long long BL = (long long)B_ * L_;      // 4096
  const long long BTD = BT * D_;
  const long long BLD = BL * D_;

  char* wp = (char*)d_ws;
  auto carve = [&](size_t bytes) -> float* {
    float* p = (float*)wp;
    wp += (bytes + 255) & ~(size_t)255;
    return p;
  };
  float* mem  = carve(BTD * 4);
  float* K2h  = carve(BTD * 4);                 // [B,H,T,HD]
  float* V2t  = carve(BTD * 4);                 // [B,H,HD,T]
  float* pad  = carve(BT * 4);
  float* tgt  = carve(BLD * 4);
  float* bufQ = carve(BLD * 4);
  float* bufK = carve(BLD * 4);
  float* bufV = carve(BLD * 4);
  float* Qh   = carve(BLD * 4);                 // [B,H,L,HD]
  float* Kh   = carve(BLD * 4);
  float* Vt   = carve(BLD * 4);                 // [B,H,HD,L]
  float* Oh   = carve(BLD * 4);
  float* ybuf = carve(BLD * 4);
  float* y2   = carve(BLD * 4);
  float* ffn  = carve(BL * DFF_ * 4);
  float* S    = carve((long long)B_ * H_ * L_ * T_ * 4);   // big enough for both attns
  float* kc   = S;                               // temp reuse before attention runs
  float* vc   = S + BTD;

  const int BHL = B_ * H_ * L_;
  const int BH  = B_ * H_;

  // ---- memory = enc @ inp_w^T + inp_b ;  pad mask ----
  gemm(stream, enc, inp_w, inp_b, mem, (int)BT, D_, D_, D_, D_, D_, 0, 0, 0, 1, 0);
  pad_mask_k<<<dim3((int)BT), dim3(256), 0, stream>>>(mem, pad);

  // ---- tgt = emb[targets] * sqrt(D) ----
  embed_k<<<dim3((int)(BLD / 256)), dim3(256), 0, stream>>>(tgids, emb, tgt);

  // ---- cross-attention K/V (loop invariant: shared weights) ----
  gemm(stream, mem, mha_w + (long long)D_ * D_,     mha_b + D_,     kc, (int)BT, D_, D_, D_, D_, D_, 0, 0, 0, 1, 0);
  gemm(stream, mem, mha_w + (long long)2 * D_ * D_, mha_b + 2 * D_, vc, (int)BT, D_, D_, D_, D_, D_, 0, 0, 0, 1, 0);
  split_kv_mem_k<<<dim3((int)(BTD / 256)), dim3(256), 0, stream>>>(kc, vc, K2h, V2t);

  for (int layer = 0; layer < NL_; ++layer) {
    // ===== self-attention with RoPE =====
    gemm(stream, tgt, qw, qb, bufQ, (int)BL, D_, D_, D_, D_, D_, 0, 0, 0, 1, 0);
    gemm(stream, tgt, kw, kb, bufK, (int)BL, D_, D_, D_, D_, D_, 0, 0, 0, 1, 0);
    gemm(stream, tgt, vw, vb, bufV, (int)BL, D_, D_, D_, D_, D_, 0, 0, 0, 1, 0);
    rope_split_k<<<dim3((int)(BLD / 256)), dim3(256), 0, stream>>>(bufQ, bufK, bufV, Qh, Kh, Vt);
    // S = Qh @ Kh^T  (batched over B*H)
    gemm(stream, Qh, Kh, nullptr, S, L_, L_, HD_, HD_, HD_, L_,
         (long long)L_ * HD_, (long long)L_ * HD_, (long long)L_ * L_, BH, 0);
    softmax_self_k<<<dim3(BHL), dim3(128), 0, stream>>>(S);
    // Oh = P @ V  (B operand = Vt [HD,L])
    gemm(stream, S, Vt, nullptr, Oh, L_, HD_, L_, L_, L_, HD_,
         (long long)L_ * L_, (long long)HD_ * L_, (long long)L_ * HD_, BH, 0);
    merge_heads_k<<<dim3((int)(BLD / 256)), dim3(256), 0, stream>>>(Oh, ybuf);
    gemm(stream, ybuf, ow, ob, y2, (int)BL, D_, D_, D_, D_, D_, 0, 0, 0, 1, 0);
    add_ln_k<<<dim3((int)BL), dim3(256), 0, stream>>>(tgt, y2, ln1_g, ln1_b);

    // ===== cross-attention =====
    gemm(stream, tgt, mha_w, mha_b, bufQ, (int)BL, D_, D_, D_, D_, D_, 0, 0, 0, 1, 0);
    split_q_k<<<dim3((int)(BLD / 256)), dim3(256), 0, stream>>>(bufQ, Qh);
    gemm(stream, Qh, K2h, nullptr, S, L_, T_, HD_, HD_, HD_, T_,
         (long long)L_ * HD_, (long long)T_ * HD_, (long long)L_ * T_, BH, 0);
    softmax_cross_k<<<dim3(BHL), dim3(256), 0, stream>>>(S, pad);
    gemm(stream, S, V2t, nullptr, Oh, L_, HD_, T_, T_, T_, HD_,
         (long long)L_ * T_, (long long)HD_ * T_, (long long)L_ * HD_, BH, 0);
    merge_heads_k<<<dim3((int)(BLD / 256)), dim3(256), 0, stream>>>(Oh, ybuf);
    gemm(stream, ybuf, mha_ow, mha_ob, y2, (int)BL, D_, D_, D_, D_, D_, 0, 0, 0, 1, 0);
    add_ln_k<<<dim3((int)BL), dim3(256), 0, stream>>>(tgt, y2, ln2_g, ln2_b);

    // ===== FFN (exact GELU fused into first GEMM) =====
    gemm(stream, tgt, w1, b1, ffn, (int)BL, DFF_, D_, D_, D_, DFF_, 0, 0, 0, 1, 1);
    gemm(stream, ffn, w2, b2, y2, (int)BL, D_, DFF_, DFF_, DFF_, D_, 0, 0, 0, 1, 0);
    add_ln_k<<<dim3((int)BL), dim3(256), 0, stream>>>(tgt, y2, ln3_g, ln3_b);
  }

  // ---- logits = tgt @ out_w^T + out_b  -> d_out ; then append tgt ----
  float* logits = (float*)d_out;
  gemm(stream, tgt, out_w, out_b, logits, (int)BL, V_, D_, D_, D_, V_, 0, 0, 0, 1, 0);
  hipMemcpyAsync(logits + BL * (long long)V_, tgt, BLD * 4,
                 hipMemcpyDeviceToDevice, stream);
}
